// Sage_52518860096077
// MI455X (gfx1250) — matrix operand order repaired and checked by reading, not compiled, not run on hardware
//
#include <hip/hip_runtime.h>
#include <hip/hip_bf16.h>

// ---------------------------------------------------------------------------
// GraphSAGE 2-layer forward for MI455X (gfx1250).
// Scatter phase is memory/L2-atomic bound (~1.2 GB traffic @ 23.3 TB/s);
// GEMMs (<9 GFLOP) run on v_wmma_f32_16x16x32_bf16 with f32 accumulate.
// ---------------------------------------------------------------------------

#define N_NODES   50000
#define N_EDGES   800000
#define F_INN     128
#define HIDDEN_D  256
#define N_CLASSES 40
#define NC_PAD    48      // 3 tiles of 16

typedef __attribute__((ext_vector_type(16))) __bf16 v16bf;
typedef __attribute__((ext_vector_type(8)))  float  v8f;

union FragAB {
    v16bf v;
    unsigned int u[8];
};

__device__ __forceinline__ unsigned short f2bf(float x) {
    unsigned int u = __float_as_uint(x);
    unsigned int r = u + 0x7FFFu + ((u >> 16) & 1u);   // round-to-nearest-even
    return (unsigned short)(r >> 16);
}

// A fragment (16x32 bf16) gather from an LDS tile stored row-major [rows][rowStride]
// Per ISA 7.12.2: lane L -> row M = L&15; VGPR v, half h ->
//   K = (v>=4 ? 16 : 0) + (L>>4)*8 + (v&3)*2 + h   (pairs of K contiguous -> b32 loads)
__device__ __forceinline__ v16bf load_a_frag(const unsigned short* lds, int rowStride,
                                             int mBase, int kBase, int lane) {
    FragAB f;
    const unsigned short* row = lds + (mBase + (lane & 15)) * rowStride;
    int g8 = (lane >> 4) << 3;
#pragma unroll
    for (int v = 0; v < 8; ++v) {
        int k = kBase + ((v & 4) ? 16 : 0) + g8 + ((v & 3) << 1);
        f.u[v] = *(const unsigned int*)(row + k);
    }
    return f.v;
}

// B fragment: pre-packed layout, 16 bf16 (32 B) contiguous per lane -> 2x b128 loads.
__device__ __forceinline__ v16bf load_b_frag(const unsigned short* packed,
                                             int ntile, int kchunk, int kChunks, int lane) {
    const unsigned short* p = packed + ((((ntile * kChunks) + kchunk) << 5) + lane) * 16;
    const uint4* q = (const uint4*)p;
    uint4 a = q[0], b = q[1];
    FragAB f;
    f.u[0] = a.x; f.u[1] = a.y; f.u[2] = a.z; f.u[3] = a.w;
    f.u[4] = b.x; f.u[5] = b.y; f.u[6] = b.z; f.u[7] = b.w;
    return f.v;
}

__device__ __forceinline__ v8f wmma_bf16(v16bf a, v16bf b, v8f c) {
    return __builtin_amdgcn_wmma_f32_16x16x32_bf16(false, a, false, b,
                                                   (short)0, c, false, false);
}

// ---------------------------------------------------------------------------

__global__ __launch_bounds__(256) void zero_kernel(float* p, int n) {
    int stride = gridDim.x * blockDim.x;
    for (int i = blockIdx.x * blockDim.x + threadIdx.x; i < n; i += stride) p[i] = 0.0f;
}

// Repack W[K][Nsrc] (f32 row-major) into per-lane WMMA B fragments (bf16).
// packed[ntile][kchunk][lane][e], k = kchunk*32 + (e&8)*2 + (lane>>4)*8 + (e&7), n = ntile*16 + (lane&15)
__global__ __launch_bounds__(256) void pack_weight(const float* __restrict__ W,
                                                   unsigned short* __restrict__ out,
                                                   int K, int Nsrc, int Ntiles) {
    int idx = blockIdx.x * blockDim.x + threadIdx.x;
    int kChunks = K >> 5;
    int total = Ntiles * kChunks * 32 * 16;
    if (idx >= total) return;
    int e      = idx & 15;
    int lane   = (idx >> 4) & 31;
    int kchunk = (idx >> 9) % kChunks;
    int ntile  = (idx >> 9) / kChunks;
    int k = (kchunk << 5) + ((e & 8) << 1) + ((lane >> 4) << 3) + (e & 7);
    int n = (ntile << 4) + (lane & 15);
    float val = (n < Nsrc) ? W[k * Nsrc + n] : 0.0f;
    out[idx] = f2bf(val);
}

__global__ __launch_bounds__(256) void degree_kernel(const int* __restrict__ dst,
                                                     float* __restrict__ deg, int nE) {
    int e = blockIdx.x * blockDim.x + threadIdx.x;
    if (e < nE) atomicAdd(&deg[dst[e]], 1.0f);
}

__global__ __launch_bounds__(256) void invdeg_kernel(const float* __restrict__ deg,
                                                     float* __restrict__ inv, int n) {
    int i = blockIdx.x * blockDim.x + threadIdx.x;
    if (i < n) inv[i] = 1.0f / fmaxf(deg[i], 1.0f);
}

// One thread per (edge, float4 feature group): float4 gather + 4 f32 atomics
// (accumulators fit in the 192 MB L2, so atomics resolve at L2 rate).
__global__ __launch_bounds__(256) void scatter_kernel(const int* __restrict__ src,
                                                      const int* __restrict__ dst,
                                                      const float* __restrict__ feat,
                                                      float* __restrict__ agg,
                                                      int nE, int F4) {
    int gid = blockIdx.x * blockDim.x + threadIdx.x;
    if (gid >= nE * F4) return;
    int f = gid % F4;
    int e = gid / F4;
    int F = F4 << 2;
    float4 v = ((const float4*)(feat + (size_t)src[e] * F))[f];
    float* a = agg + (size_t)dst[e] * F + (f << 2);
    atomicAdd(a + 0, v.x);
    atomicAdd(a + 1, v.y);
    atomicAdd(a + 2, v.z);
    atomicAdd(a + 3, v.w);
}

// Layer 1: h = relu(mean @ W1_l + x @ W1_r + b1).  M-block = 32 rows, 8 waves,
// each wave owns 2 N-tiles x 2 M-subtiles; K = 128 -> 4 chunks, 2 WMMA per chunk.
__global__ __launch_bounds__(256) void sage1_kernel(const float* __restrict__ agg,
                                                    const float* __restrict__ x,
                                                    const float* __restrict__ inv_deg,
                                                    const unsigned short* __restrict__ w1l_p,
                                                    const unsigned short* __restrict__ w1r_p,
                                                    const float* __restrict__ b1,
                                                    float* __restrict__ h) {
    __shared__ unsigned short sMean[32 * F_INN];
    __shared__ unsigned short sX[32 * F_INN];
    int tid = threadIdx.x;
    int mBlock = blockIdx.x << 5;

    for (int i = tid; i < 32 * F_INN; i += 256) {
        int r = i >> 7, c = i & (F_INN - 1);
        int grow = mBlock + r;
        float xm = 0.0f, xx = 0.0f;
        if (grow < N_NODES) {
            float id = inv_deg[grow];
            xm = agg[(size_t)grow * F_INN + c] * id;
            xx = x[(size_t)grow * F_INN + c];
        }
        sMean[i] = f2bf(xm);
        sX[i]    = f2bf(xx);
    }
    __syncthreads();

    int wave = tid >> 5, lane = tid & 31;
#pragma unroll
    for (int t = 0; t < 2; ++t) {
        int ntile = (wave << 1) + t;
#pragma unroll
        for (int sm = 0; sm < 2; ++sm) {
            v8f acc = {};
#pragma unroll
            for (int kc = 0; kc < 4; ++kc) {
                v16bf am = load_a_frag(sMean, F_INN, sm << 4, kc << 5, lane);
                v16bf bl = load_b_frag(w1l_p, ntile, kc, 4, lane);
                acc = wmma_bf16(am, bl, acc);
                v16bf ax = load_a_frag(sX, F_INN, sm << 4, kc << 5, lane);
                v16bf br = load_b_frag(w1r_p, ntile, kc, 4, lane);
                acc = wmma_bf16(ax, br, acc);
            }
            int n = (ntile << 4) + (lane & 15);
            float bias = b1[n];
#pragma unroll
            for (int j = 0; j < 8; ++j) {
                int m = mBlock + (sm << 4) + j + ((lane >> 4) << 3);
                if (m < N_NODES) {
                    float v = acc[j] + bias;
                    h[(size_t)m * HIDDEN_D + n] = v > 0.0f ? v : 0.0f;
                }
            }
        }
    }
}

// Layer 2: logits = mean2 @ W2_l + h @ W2_r + b2 (N padded 40->48, 3 N-tiles).
// K = 256 -> 8 chunks. Waves 0..5 compute (3 ntiles x 2 sub-M), all waves stage.
__global__ __launch_bounds__(256) void sage2_kernel(const float* __restrict__ agg,
                                                    const float* __restrict__ h,
                                                    const float* __restrict__ inv_deg,
                                                    const unsigned short* __restrict__ w2l_p,
                                                    const unsigned short* __restrict__ w2r_p,
                                                    const float* __restrict__ b2,
                                                    float* __restrict__ logits) {
    __shared__ unsigned short sMean[32 * HIDDEN_D];
    __shared__ unsigned short sH[32 * HIDDEN_D];
    int tid = threadIdx.x;
    int mBlock = blockIdx.x << 5;

    for (int i = tid; i < 32 * HIDDEN_D; i += 256) {
        int r = i >> 8, c = i & (HIDDEN_D - 1);
        int grow = mBlock + r;
        float xm = 0.0f, xh = 0.0f;
        if (grow < N_NODES) {
            float id = inv_deg[grow];
            xm = agg[(size_t)grow * HIDDEN_D + c] * id;
            xh = h[(size_t)grow * HIDDEN_D + c];
        }
        sMean[i] = f2bf(xm);
        sH[i]    = f2bf(xh);
    }
    __syncthreads();

    int wave = tid >> 5, lane = tid & 31;
    if (wave < 6) {
        int ntile = wave % 3;
        int sm    = wave / 3;
        v8f acc = {};
#pragma unroll
        for (int kc = 0; kc < 8; ++kc) {
            v16bf am = load_a_frag(sMean, HIDDEN_D, sm << 4, kc << 5, lane);
            v16bf bl = load_b_frag(w2l_p, ntile, kc, 8, lane);
            acc = wmma_bf16(am, bl, acc);
            v16bf ah = load_a_frag(sH, HIDDEN_D, sm << 4, kc << 5, lane);
            v16bf br = load_b_frag(w2r_p, ntile, kc, 8, lane);
            acc = wmma_bf16(ah, br, acc);
        }
        int n = (ntile << 4) + (lane & 15);
        if (n < N_CLASSES) {
            float bias = b2[n];
#pragma unroll
            for (int j = 0; j < 8; ++j) {
                int m = mBlock + (sm << 4) + j + ((lane >> 4) << 3);
                if (m < N_NODES)
                    logits[(size_t)m * N_CLASSES + n] = acc[j] + bias;
            }
        }
    }
}

// One wave32 per node; lane handles classes (lane) and (lane+32).
__global__ __launch_bounds__(256) void logsoftmax_kernel(const float* __restrict__ logits,
                                                         float* __restrict__ out) {
    int node = (blockIdx.x << 3) + (threadIdx.x >> 5);
    int lane = threadIdx.x & 31;
    if (node >= N_NODES) return;
    const float* row = logits + (size_t)node * N_CLASSES;
    float v0 = row[lane];                                           // lane < 32 < 40
    float v1 = (lane + 32 < N_CLASSES) ? row[lane + 32] : -INFINITY;
    float m = fmaxf(v0, v1);
#pragma unroll
    for (int off = 16; off; off >>= 1) m = fmaxf(m, __shfl_xor(m, off, 32));
    float s = __expf(v0 - m) + ((lane + 32 < N_CLASSES) ? __expf(v1 - m) : 0.0f);
#pragma unroll
    for (int off = 16; off; off >>= 1) s += __shfl_xor(s, off, 32);
    float ls = __logf(s);
    out[(size_t)node * N_CLASSES + lane] = v0 - m - ls;
    if (lane + 32 < N_CLASSES)
        out[(size_t)node * N_CLASSES + lane + 32] = v1 - m - ls;
}

// ---------------------------------------------------------------------------
// Workspace layout (float units). Atomically-accumulated buffers first, so one
// zero_kernel launch resets everything that must be reset per replay.
#define OFF_DEG     0
#define OFF_AGG1    50048                                    // 50000*128
#define OFF_AGG2    (OFF_AGG1 + N_NODES * F_INN)             // 50000*256
#define ZERO_COUNT  (OFF_AGG2 + N_NODES * HIDDEN_D)
#define OFF_INV     ZERO_COUNT
#define OFF_H       (OFF_INV + 50048)
#define OFF_LOGITS  (OFF_H + N_NODES * HIDDEN_D)
#define OFF_PACK_F  (OFF_LOGITS + N_NODES * N_CLASSES)
#define PK_W1       (16 * 4 * 32 * 16)                       // 32768 bf16 each
#define PK_W2       (3 * 8 * 32 * 16)                        // 12288 bf16 each

extern "C" void kernel_launch(void* const* d_in, const int* in_sizes, int n_in,
                              void* d_out, int out_size, void* d_ws, size_t ws_size,
                              hipStream_t stream) {
    const float* x    = (const float*)d_in[0];
    const int*   ei   = (const int*)d_in[1];     // [2, N_EDGES]: src row then dst row
    const float* W1l  = (const float*)d_in[2];
    const float* W1r  = (const float*)d_in[3];
    const float* b1   = (const float*)d_in[4];
    const float* W2l  = (const float*)d_in[5];
    const float* W2r  = (const float*)d_in[6];
    const float* b2   = (const float*)d_in[7];
    float*       out  = (float*)d_out;

    const int* src = ei;
    const int* dst = ei + N_EDGES;

    float* ws      = (float*)d_ws;
    float* deg     = ws + OFF_DEG;
    float* agg1    = ws + OFF_AGG1;
    float* agg2    = ws + OFF_AGG2;
    float* inv     = ws + OFF_INV;
    float* h       = ws + OFF_H;
    float* logits  = ws + OFF_LOGITS;
    unsigned short* pk   = (unsigned short*)(ws + OFF_PACK_F);
    unsigned short* w1lp = pk;
    unsigned short* w1rp = pk + PK_W1;
    unsigned short* w2lp = pk + 2 * PK_W1;
    unsigned short* w2rp = pk + 2 * PK_W1 + PK_W2;

    // 1) zero the atomic accumulators (deg + agg1 + agg2)
    zero_kernel<<<4096, 256, 0, stream>>>(ws, ZERO_COUNT);

    // 2) repack weights into WMMA B-fragment layout (bf16)
    pack_weight<<<(PK_W1 + 255) / 256, 256, 0, stream>>>(W1l, w1lp, F_INN, HIDDEN_D, 16);
    pack_weight<<<(PK_W1 + 255) / 256, 256, 0, stream>>>(W1r, w1rp, F_INN, HIDDEN_D, 16);
    pack_weight<<<(PK_W2 + 255) / 256, 256, 0, stream>>>(W2l, w2lp, HIDDEN_D, N_CLASSES, 3);
    pack_weight<<<(PK_W2 + 255) / 256, 256, 0, stream>>>(W2r, w2rp, HIDDEN_D, N_CLASSES, 3);

    // 3) degrees -> inv_deg
    degree_kernel<<<(N_EDGES + 255) / 256, 256, 0, stream>>>(dst, deg, N_EDGES);
    invdeg_kernel<<<(N_NODES + 255) / 256, 256, 0, stream>>>(deg, inv, N_NODES);

    // 4) layer 1: scatter + fused dual-GEMM + bias + relu
    scatter_kernel<<<(N_EDGES * (F_INN / 4) + 255) / 256, 256, 0, stream>>>(
        src, dst, x, agg1, N_EDGES, F_INN / 4);
    sage1_kernel<<<(N_NODES + 31) / 32, 256, 0, stream>>>(agg1, x, inv, w1lp, w1rp, b1, h);

    // 5) layer 2: scatter + fused dual-GEMM + bias
    scatter_kernel<<<(N_EDGES * (HIDDEN_D / 4) + 255) / 256, 256, 0, stream>>>(
        src, dst, h, agg2, N_EDGES, HIDDEN_D / 4);
    sage2_kernel<<<(N_NODES + 31) / 32, 256, 0, stream>>>(agg2, h, inv, w2lp, w2rp, b2, logits);

    // 6) log_softmax -> output
    logsoftmax_kernel<<<(N_NODES + 7) / 8, 256, 0, stream>>>(logits, out);
}
